// GCNAggregator_41755672051923
// MI455X (gfx1250) — matrix operand order repaired and checked by reading, not compiled, not run on hardware
//
#include <hip/hip_runtime.h>
#include <hip/hip_bf16.h>

typedef __attribute__((ext_vector_type(16))) _Float16 v16h;
typedef __attribute__((ext_vector_type(8)))  _Float16 v8h;
typedef __attribute__((ext_vector_type(8)))  float    v8f;

#define NEIGH_K   32
#define EMB_D     128
#define SLOTS     33                  // 32 neighbors + self
#define ROWS_TILE 16
#define KSLOTS    (ROWS_TILE * SLOTS) // 528
#define KPAD      544                 // 17 chunks of 32
#define NCHUNK    (KPAD / 32)

// ---------------- kernel 0: zero the column-degree counters ----------------
__global__ void gcn_zero_counts(int* __restrict__ cnt, int m) {
    int i = blockIdx.x * blockDim.x + threadIdx.x;
    if (i < m) cnt[i] = 0;
}

// ---------------- kernel 1: per-row dedupe + column degree counts ----------
// One wave32 per row. owner(k) = neighbor k is the first occurrence of its
// value among the 32 neighbor slots; self is an owner iff it matches no
// neighbor. Each unique column of the row has exactly one owner -> exact
// set-semantics row/col degrees as in the reference.
__global__ void gcn_count(const int* __restrict__ nodes,
                          const int* __restrict__ neigh,
                          int*       __restrict__ cnt,
                          unsigned*  __restrict__ omask,
                          unsigned*  __restrict__ meta,
                          int n) {
    int lane = threadIdx.x & 31;
    int row  = (blockIdx.x * blockDim.x + threadIdx.x) >> 5;
    if (row >= n) return;                       // uniform per wave

    int v    = neigh[row * NEIGH_K + lane];
    int self = nodes[row];

    unsigned dup = 0;
    #pragma unroll
    for (int j = 0; j < 31; ++j) {
        int ov = __shfl(v, j);
        dup |= (j < lane && ov == v) ? 1u : 0u;
    }
    int owner = dup ? 0 : 1;

    unsigned long long selfmatch = __ballot(v == self);
    unsigned om = (unsigned)__ballot(owner != 0);
    int selfOwner = (selfmatch == 0ull) ? 1 : 0;
    int rowdeg = __popc(om) + selfOwner;

    if (owner) atomicAdd(&cnt[v], 1);
    if (lane == 0) {
        if (selfOwner) atomicAdd(&cnt[self], 1);
        omask[row] = om;
        meta[row]  = (unsigned)rowdeg | ((unsigned)selfOwner << 16);
    }
}

// ---------------- kernel 2: gather-GEMM aggregation via WMMA ---------------
// One 256-thread block per 16-row tile. K-dim = 16 rows x 33 slots = 528
// (padded to 544 = 17 chunks of 32). The block-diagonal A tile is
// materialized densely in LDS so the MMA loop does only contiguous
// ds_load_b128 reads (no predication, no divides), B is gathered from the
// embedding matrix with 64B-coalesced loads, f32 accumulation in WMMA.
__global__ void __launch_bounds__(256)
gcn_aggregate(const int*      __restrict__ nodes,
              const int*      __restrict__ neigh,
              const float*    __restrict__ embed,
              const int*      __restrict__ cnt,
              const unsigned* __restrict__ omask,
              const unsigned* __restrict__ meta,
              float*          __restrict__ out) {
    __shared__ __align__(16) _Float16 sA[ROWS_TILE][KPAD]; // dense A tile (17.4 KB)
    __shared__ __align__(16) int      sIdx[KPAD];          // gather indices

    const int tile = blockIdx.x;
    const int tid  = threadIdx.x;

    // ---- stage 1a: zero the A tile (vectorized as u32) ----
    {
        unsigned* za = (unsigned*)&sA[0][0];
        #pragma unroll
        for (int t = tid; t < ROWS_TILE * KPAD / 2; t += 256) za[t] = 0u;
    }
    __syncthreads();

    // ---- stage 1b: scatter weights + slot indices ----
    for (int t = tid; t < KPAD; t += 256) {
        int idx = 0;
        if (t < KSLOTS) {
            int m   = t / SLOTS;
            int k   = t - m * SLOTS;
            int row = tile * ROWS_TILE + m;
            int ci, own;
            if (k < NEIGH_K) {
                ci  = neigh[row * NEIGH_K + k];
                own = (omask[row] >> k) & 1;
            } else {
                ci  = nodes[row];
                own = (meta[row] >> 16) & 1;
            }
            idx = ci;
            if (own) {
                float rd = (float)(meta[row] & 0xFFFFu);
                float cd = (float)cnt[ci];          // >= 1 (owner exists)
                sA[m][t] = (_Float16)(rsqrtf(rd) * rsqrtf(cd));
            }
        }
        sIdx[t] = idx;
    }
    __syncthreads();

    const int lane = tid & 31;
    const int wv   = tid >> 5;          // column tile 0..7 (D = 128)
    const int h    = lane >> 4;         // lane half
    const int ncol = lane & 15;
    const int col  = (wv << 4) + ncol;  // global output column

    const _Float16* aRow = &sA[ncol][0];     // A row owned by this lane
    const float*    eC   = embed + col;      // column-shifted embed base

    v8f acc = {};

    for (int c = 0; c < NCHUNK; ++c) {
        const int base = c << 5;

        // A fragment: elements 0..7 -> slots base+16h .. +7,
        //             elements 8..15 -> slots base+16h+8 .. +15  (2x b128)
        v8h alo = *(const v8h*)(aRow + base + (h << 4));
        v8h ahi = *(const v8h*)(aRow + base + (h << 4) + 8);
        v16h a  = __builtin_shufflevector(alo, ahi,
                    0,1,2,3,4,5,6,7,8,9,10,11,12,13,14,15);

        // B slot indices: slots base+16h .. +15 contiguous (4x b128)
        const int* ip = &sIdx[base + (h << 4)];
        int4 i0 = *(const int4*)(ip + 0);
        int4 i1 = *(const int4*)(ip + 4);
        int4 i2 = *(const int4*)(ip + 8);
        int4 i3 = *(const int4*)(ip + 12);

        // B fragment: gather + f32->f16 convert
        v16h b;
        b[ 0] = (_Float16)eC[i0.x * EMB_D];
        b[ 1] = (_Float16)eC[i0.y * EMB_D];
        b[ 2] = (_Float16)eC[i0.z * EMB_D];
        b[ 3] = (_Float16)eC[i0.w * EMB_D];
        b[ 4] = (_Float16)eC[i1.x * EMB_D];
        b[ 5] = (_Float16)eC[i1.y * EMB_D];
        b[ 6] = (_Float16)eC[i1.z * EMB_D];
        b[ 7] = (_Float16)eC[i1.w * EMB_D];
        b[ 8] = (_Float16)eC[i2.x * EMB_D];
        b[ 9] = (_Float16)eC[i2.y * EMB_D];
        b[10] = (_Float16)eC[i2.z * EMB_D];
        b[11] = (_Float16)eC[i2.w * EMB_D];
        b[12] = (_Float16)eC[i3.x * EMB_D];
        b[13] = (_Float16)eC[i3.y * EMB_D];
        b[14] = (_Float16)eC[i3.z * EMB_D];
        b[15] = (_Float16)eC[i3.w * EMB_D];

        acc = __builtin_amdgcn_wmma_f32_16x16x32_f16(
            /*neg_a=*/false, a, /*neg_b=*/false, b,
            /*c_mod=*/(short)0, acc, /*reuse_a=*/false, /*reuse_b=*/false);
    }

    // C/D layout: VGPR r -> row (r + 8*h) of the tile, col = lane&15.
    #pragma unroll
    for (int r = 0; r < 8; ++r) {
        int row = tile * ROWS_TILE + r + (h << 3);
        out[row * EMB_D + col] = acc[r];
    }
}

// ---------------------------------------------------------------------------
extern "C" void kernel_launch(void* const* d_in, const int* in_sizes, int n_in,
                              void* d_out, int out_size, void* d_ws, size_t ws_size,
                              hipStream_t stream) {
    (void)n_in; (void)out_size; (void)ws_size;

    const int*   nodes = (const int*)d_in[0];
    const int*   neigh = (const int*)d_in[1];
    const float* embed = (const float*)d_in[2];
    float*       out   = (float*)d_out;

    const int N = in_sizes[0];                 // 4096
    const int M = in_sizes[2] / EMB_D;         // 16384

    // Workspace layout: cnt[M] | omask[N] | meta[N]  (~96 KB)
    char* ws = (char*)d_ws;
    int*      cnt = (int*)ws;
    unsigned* om  = (unsigned*)(ws + (size_t)M * sizeof(int));
    unsigned* mt  = (unsigned*)(ws + (size_t)M * sizeof(int) + (size_t)N * sizeof(unsigned));

    gcn_zero_counts<<<(M + 255) / 256, 256, 0, stream>>>(cnt, M);
    gcn_count<<<(N * 32 + 255) / 256, 256, 0, stream>>>(nodes, neigh, cnt, om, mt, N);
    gcn_aggregate<<<N / ROWS_TILE, 256, 0, stream>>>(nodes, neigh, embed, cnt, om, mt, out);
}
